// DifferentialEntropyRegularization_62191126446270
// MI455X (gfx1250) — matrix operand order repaired and checked by reading, not compiled, not run on hardware
//
#include <hip/hip_runtime.h>
#include <hip/hip_bf16.h>
#include <math.h>

// MI455X / gfx1250: wave32, WMMA 16x16x32 f16 -> f32.
typedef __attribute__((ext_vector_type(16))) _Float16 v16h;
typedef __attribute__((ext_vector_type(8)))  _Float16 v8h;
typedef __attribute__((ext_vector_type(8)))  float    v8f;

#define N_ROWS 8192
#define DIM    512
#define KCH    (DIM / 32)        // 16 K-chunks of 32 per tile
#define LDS_PAD 8                // halves; breaks 64-bank alignment of 1024B rows
#define BROW   (DIM + LDS_PAD)   // 520 halves per LDS row (1040 B, 16B-aligned)
#define CSPLIT 4                 // column splits (blockIdx.y)
#define CTILES (N_ROWS / CSPLIT / 16)   // 128 column tiles per split

union AB16 { v16h v; v8h h[2]; };

// ---------------- K1: fp32 -> fp16 convert (x becomes L2-resident) ----------
__global__ void cvt_f32_to_f16(const float* __restrict__ x,
                               _Float16* __restrict__ xh, int n) {
  int i = blockIdx.x * blockDim.x + threadIdx.x;
  if (i < n) xh[i] = (_Float16)x[i];
}

// ---------------- K2: fused X @ X^T + masked row argmax (partial) -----------
// Block = 4 waves; each wave owns TWO register-resident 16-row A slabs
// (32 rows, ~256 VGPRs) so every B tile pulled from LDS feeds 2 WMMAs,
// halving LDS traffic per FLOP. Block covers 128 rows x 2048 columns.
__global__ __launch_bounds__(128)
void nn_argmax_wmma(const _Float16* __restrict__ xh,
                    float* __restrict__ pmax, int* __restrict__ pidx) {
  __shared__ _Float16 Bs[16 * BROW];

  const int tid  = threadIdx.x;
  const int wave = tid >> 5;
  const int lane = tid & 31;
  const int n    = lane & 15;   // tile column / A-row within slab
  const int hi   = lane >> 4;   // half-wave selector
  const int i0   = blockIdx.x * 128 + wave * 32;   // slab0 rows i0.., slab1 i0+16..
  const int jbase = blockIdx.y * (N_ROWS / CSPLIT);

  // A 16x32 f16 layout (ISA 7.12.2): lane m<16 holds K 0..7 & 16..23 of row m,
  // lane m+16 holds K 8..15 & 24..31. Full K=512 slabs loaded once.
  AB16 a0[KCH], a1[KCH];
  {
    const _Float16* r0 = xh + (size_t)(i0 + n) * DIM;
    const _Float16* r1 = xh + (size_t)(i0 + 16 + n) * DIM;
#pragma unroll
    for (int kc = 0; kc < KCH; ++kc) {
      a0[kc].h[0] = *(const v8h*)(r0 + kc * 32 + hi * 8);
      a0[kc].h[1] = *(const v8h*)(r0 + kc * 32 + 16 + hi * 8);
      a1[kc].h[0] = *(const v8h*)(r1 + kc * 32 + hi * 8);
      a1[kc].h[1] = *(const v8h*)(r1 + kc * 32 + 16 + hi * 8);
    }
  }

  float best0[8], best1[8];
  int   bidx0[8], bidx1[8];
#pragma unroll
  for (int v = 0; v < 8; ++v) {
    best0[v] = -INFINITY; bidx0[v] = 0x7fffffff;
    best1[v] = -INFINITY; bidx1[v] = 0x7fffffff;
  }

  for (int jt = 0; jt < CTILES; ++jt) {
    const int j0 = jbase + jt * 16;

    __syncthreads();  // all waves done reading previous B slab
    // Cooperative B-slab load: 16 rows x 512 halves = 1024 chunks of 8 halves.
#pragma unroll
    for (int q = 0; q < 8; ++q) {
      int c   = tid + q * 128;
      int r   = c >> 6;          // 64 chunks per row
      int off = (c & 63) * 8;
      *(v8h*)(&Bs[r * BROW + off]) =
          *(const v8h*)(xh + (size_t)(j0 + r) * DIM + off);
    }
    __syncthreads();

    if (jt + 1 < CTILES)  // warm next slab (global_prefetch_b8)
      __builtin_prefetch(xh + (size_t)(j0 + 16 + n) * DIM, 0, 0);

    // B 32x16 f16 layout: lanes 0-15 hold K 0..7 & 16..23 of column n, lanes
    // 16-31 hold K 8..15 & 24..31 (matches A).
    v8f c0 = {}, c1 = {};
    const _Float16* brow = &Bs[n * BROW + hi * 8];
#pragma unroll
    for (int kc = 0; kc < KCH; ++kc) {
      AB16 b;
      b.h[0] = *(const v8h*)(brow + kc * 32);
      b.h[1] = *(const v8h*)(brow + kc * 32 + 16);
      c0 = __builtin_amdgcn_wmma_f32_16x16x32_f16(
          false, a0[kc].v, false, b.v, (short)0, c0, false, false);
      c1 = __builtin_amdgcn_wmma_f32_16x16x32_f16(
          false, a1[kc].v, false, b.v, (short)0, c1, false, false);
    }

    // D layout: lane holds column j0+n; VGPR v -> slab row 8*hi + v.
    const int col = j0 + n;
#pragma unroll
    for (int v = 0; v < 8; ++v) {
      const int row0 = i0 + 8 * hi + v;
      const int row1 = row0 + 16;
      const float v0 = c0[v], v1 = c1[v];
      const bool t0 = (col != row0) & (v0 > best0[v]);   // excl. diagonal
      const bool t1 = (col != row1) & (v1 > best1[v]);
      best0[v] = t0 ? v0 : best0[v];  bidx0[v] = t0 ? col : bidx0[v];
      best1[v] = t1 ? v1 : best1[v];  bidx1[v] = t1 ? col : bidx1[v];
    }
  }

  // Butterfly (max, lowest-index-on-tie) across the 16 lanes of each half.
#pragma unroll
  for (int v = 0; v < 8; ++v) {
    float val0 = best0[v], val1 = best1[v];
    int   idx0 = bidx0[v], idx1 = bidx1[v];
#pragma unroll
    for (int m = 1; m < 16; m <<= 1) {
      float o0 = __shfl_xor(val0, m, 32), o1 = __shfl_xor(val1, m, 32);
      int   q0 = __shfl_xor(idx0, m, 32), q1 = __shfl_xor(idx1, m, 32);
      if (o0 > val0 || (o0 == val0 && q0 < idx0)) { val0 = o0; idx0 = q0; }
      if (o1 > val1 || (o1 == val1 && q1 < idx1)) { val1 = o1; idx1 = q1; }
    }
    if (n == 0) {
      const size_t base = (size_t)blockIdx.y * N_ROWS;
      pmax[base + i0 + 8 * hi + v]      = val0;
      pidx[base + i0 + 8 * hi + v]      = idx0;
      pmax[base + i0 + 16 + 8 * hi + v] = val1;
      pidx[base + i0 + 16 + 8 * hi + v] = idx1;
    }
  }
}

// ------- K3: merge column-split partials, then log(||x_i - x_nn + eps|| ) ---
__global__ void row_log_rho(const float* __restrict__ x,
                            const float* __restrict__ pmax,
                            const int* __restrict__ pidx,
                            float* __restrict__ logs) {
  const int row  = blockIdx.x * (blockDim.x >> 5) + (threadIdx.x >> 5);
  const int lane = threadIdx.x & 31;
  if (row >= N_ROWS) return;

  // Ascending split order + lowest-index tie-break == jnp.argmax semantics.
  float bv = -INFINITY; int bi = 0;
#pragma unroll
  for (int s = 0; s < CSPLIT; ++s) {
    float v = pmax[(size_t)s * N_ROWS + row];
    int   i = pidx[(size_t)s * N_ROWS + row];
    if (v > bv || (v == bv && i < bi)) { bv = v; bi = i; }
  }

  const float* xr = x + (size_t)row * DIM;
  const float* xn = x + (size_t)bi * DIM;
  float acc = 0.f;
#pragma unroll 4
  for (int k = lane; k < DIM; k += 32) {
    float d = xr[k] - xn[k] + 1e-6f;
    acc += d * d;
  }
#pragma unroll
  for (int m = 16; m >= 1; m >>= 1) acc += __shfl_xor(acc, m, 32);
  if (lane == 0) logs[row] = logf(sqrtf(acc) + 1e-8f);
}

// ---------------- K4: deterministic -mean reduction -------------------------
__global__ void reduce_mean_neg(const float* __restrict__ logs,
                                float* __restrict__ out) {
  __shared__ float sm[256];
  float acc = 0.f;
  for (int i = threadIdx.x; i < N_ROWS; i += 256) acc += logs[i];
  sm[threadIdx.x] = acc;
  __syncthreads();
  for (int s = 128; s > 0; s >>= 1) {
    if (threadIdx.x < s) sm[threadIdx.x] += sm[threadIdx.x + s];
    __syncthreads();
  }
  if (threadIdx.x == 0) out[0] = -sm[0] / (float)N_ROWS;
}

extern "C" void kernel_launch(void* const* d_in, const int* in_sizes, int n_in,
                              void* d_out, int out_size, void* d_ws, size_t ws_size,
                              hipStream_t stream) {
  const float* x  = (const float*)d_in[0];
  float*      out = (float*)d_out;

  char* ws = (char*)d_ws;
  _Float16* xh  = (_Float16*)ws;                                  // 8 MB
  size_t off = (size_t)N_ROWS * DIM * sizeof(_Float16);
  float* pmax = (float*)(ws + off);  off += (size_t)CSPLIT * N_ROWS * sizeof(float);
  int*   pidx = (int*)(ws + off);    off += (size_t)CSPLIT * N_ROWS * sizeof(int);
  float* logs = (float*)(ws + off);

  const int total = N_ROWS * DIM;
  cvt_f32_to_f16<<<(total + 255) / 256, 256, 0, stream>>>(x, xh, total);

  dim3 grid(N_ROWS / 128, CSPLIT);
  nn_argmax_wmma<<<grid, 128, 0, stream>>>(xh, pmax, pidx);

  row_log_rho<<<N_ROWS / 8, 256, 0, stream>>>(x, pmax, pidx, logs);
  reduce_mean_neg<<<1, 256, 0, stream>>>(logs, out);
}
